// MoELayer_36026185679367
// MI455X (gfx1250) — compile-verified, matrix operand order checked by reading, not run on hardware
//
#include <hip/hip_runtime.h>
#include <hip/hip_bf16.h>

#define D_EMB 1024
#define H_DIM 4096
#define N_EXP 8
#define T_TOK 4096

typedef __bf16 bf16t;
typedef __attribute__((ext_vector_type(16))) __bf16 v16bf;
typedef __attribute__((ext_vector_type(8)))  float   v8f;
typedef __attribute__((ext_vector_type(4)))  unsigned int v4u;

union FragU { v16bf v; v4u q[2]; };

// LDS pitches (elements). Row strides in bytes: 2064, 80, 272 -> 16B aligned,
// and (stride/4) % 64 == 4 (XP, HP) or 20 (BP) banks so the 16 M-lanes of a fragment load spread
// across distinct bank groups.
#define XP 1032   // x tile pitch   (1024 + 8)
#define BP 40     // B-tile pitch   (32   + 8)  [fallback path only]
#define HP 136    // h tile pitch   (128  + 8)

// ---------------------------------------------------------------- zero ----
__global__ void k_zero(float* __restrict__ out, int* __restrict__ cnt, int n) {
    int i = blockIdx.x * blockDim.x + threadIdx.x;
    if (i < N_EXP) cnt[i] = 0;
    for (int j = i; j < n; j += gridDim.x * blockDim.x) out[j] = 0.0f;
}

// -------------------------------------------------------------- router ----
__global__ void k_router(const float* __restrict__ x,
                         const float* __restrict__ rw,
                         const float* __restrict__ rb,
                         int* __restrict__ cnt,
                         int* __restrict__ tlist,
                         float* __restrict__ glist) {
    int lane = threadIdx.x & 31;
    int wid  = threadIdx.x >> 5;
    int t    = blockIdx.x * 8 + wid;
    if (t >= T_TOK) return;

    const float* xr = x + (size_t)t * D_EMB;
    float acc[N_EXP];
#pragma unroll
    for (int e = 0; e < N_EXP; ++e) acc[e] = 0.0f;

    for (int d = lane; d < D_EMB; d += 32) {
        float xv = xr[d];
        const float* w = rw + (size_t)d * N_EXP;
#pragma unroll
        for (int e = 0; e < N_EXP; ++e) acc[e] += xv * w[e];
    }
#pragma unroll
    for (int off = 16; off > 0; off >>= 1) {
#pragma unroll
        for (int e = 0; e < N_EXP; ++e) acc[e] += __shfl_xor(acc[e], off, 32);
    }
    if (lane == 0) {
        float lg[N_EXP];
#pragma unroll
        for (int e = 0; e < N_EXP; ++e) lg[e] = acc[e] + rb[e];
        int e0 = 0; float l0 = lg[0];
#pragma unroll
        for (int e = 1; e < N_EXP; ++e) if (lg[e] > l0) { l0 = lg[e]; e0 = e; }
        int e1 = (e0 == 0) ? 1 : 0; float l1 = lg[e1];
#pragma unroll
        for (int e = 0; e < N_EXP; ++e)
            if (e != e0 && lg[e] > l1) { l1 = lg[e]; e1 = e; }
        float g0 = 1.0f / (1.0f + __expf(l1 - l0));
        float g1 = 1.0f - g0;
        int p0 = atomicAdd(&cnt[e0], 1);
        tlist[e0 * T_TOK + p0] = t;  glist[e0 * T_TOK + p0] = g0;
        int p1 = atomicAdd(&cnt[e1], 1);
        tlist[e1 * T_TOK + p1] = t;  glist[e1 * T_TOK + p1] = g1;
    }
}

// ------------------------------------------------------------- packers ----
// x: fp32 -> bf16, row-major.
__global__ void k_pack_x(const float* __restrict__ x, bf16t* __restrict__ xb,
                         int n) {
    int base = (blockIdx.x * blockDim.x + threadIdx.x) * 4;
    if (base >= n) return;
    float4 v = *(const float4*)(x + base);
    xb[base + 0] = (bf16t)v.x; xb[base + 1] = (bf16t)v.y;
    xb[base + 2] = (bf16t)v.z; xb[base + 3] = (bf16t)v.w;
}

// Pack `nmats` KxN row-major fp32 matrices into exact WMMA bf16 B-fragment
// layout: dst[mat][n/16][k/32][lane][j], j0..7 -> K klo..klo+7,
// j8..15 -> K 16+klo..16+klo+7, klo = (lane>=16)*8, N col = (lane&15).
// Each 32x16 tile becomes 1KB contiguous; a wave fetches it with two
// global_load_b128 per lane, fully coalesced.
__global__ void k_pack_b(const float* __restrict__ src, bf16t* __restrict__ dst,
                         int K, int N, int nmats) {
    long idx  = (long)blockIdx.x * blockDim.x + threadIdx.x;
    int lane  = (int)(idx & 31);
    long rest = idx >> 5;
    int ks = (int)(rest % (K / 32));  rest /= (K / 32);
    int nt = (int)(rest % (N / 16));
    long mat = rest / (N / 16);
    if (mat >= nmats) return;
    const float* s = src + mat * (long)K * N;
    bf16t* d = dst + (((mat * (N / 16) + nt) * (long)(K / 32) + ks) * 32 + lane) * 16;
    int n   = nt * 16 + (lane & 15);
    int klo = (lane >> 4) * 8;
#pragma unroll
    for (int j = 0; j < 16; ++j) {
        int k = ks * 32 + ((j < 8) ? (klo + j) : (16 + klo + (j - 8)));
        d[j] = (bf16t)s[(long)k * N + n];
    }
}

// ------------------------------------------------- ffn (packed weights) ----
// Block = 16 tokens of one expert, 8 waves. B fragments come straight from
// pre-packed global bf16 (2x b128 per lane), no B staging, no barriers in
// the K loops. Wave w: h-cols [w*16,w*16+16) in GEMM1, y-cols
// [w*128,(w+1)*128) in GEMM2 (8 persistent WMMA C tiles).
__global__ __launch_bounds__(256) void k_ffn_packed(
    const bf16t* __restrict__ xb,  const bf16t* __restrict__ w1p,
    const bf16t* __restrict__ w2p, const float* __restrict__ b1,
    const float* __restrict__ b2,  const int* __restrict__ cnt,
    const int* __restrict__ tlist, const float* __restrict__ glist,
    float* __restrict__ out) {

    __shared__ __attribute__((aligned(16))) bf16t xlds[16 * XP];  // 33.0 KB
    __shared__ __attribute__((aligned(16))) bf16t hlds[16 * HP];  //  4.4 KB

    int e    = blockIdx.x >> 8;
    int tile = blockIdx.x & 255;
    int ntok = cnt[e];
    if (tile * 16 >= ntok) return;

    int tid  = threadIdx.x;
    int lane = tid & 31, wv = tid >> 5;
    int half = lane >> 4;
    int klo  = half * 8;
    int nl   = lane & 15;

    const bf16t* w1e = w1p + (size_t)e * (H_DIM / 16) * (D_EMB / 32) * 512;
    const bf16t* w2e = w2p + (size_t)e * (D_EMB / 16) * (H_DIM / 32) * 512;

    // ---- stage x tile: pure b128 copies (already bf16) ----
    {
        int m = tid >> 4, g = tid & 15;
        int r  = tile * 16 + m;
        int rr = (r < ntok) ? r : (ntok - 1);
        int tok = tlist[e * T_TOK + rr];
        const bf16t* xr = xb + (size_t)tok * D_EMB;
#pragma unroll
        for (int q = 0; q < 8; ++q) {
            int col = q * 128 + g * 8;
            *(v4u*)&xlds[m * XP + col] = *(const v4u*)(xr + col);
        }
    }
    __syncthreads();

    v8f acc[8];
#pragma unroll
    for (int nt = 0; nt < 8; ++nt)
#pragma unroll
        for (int i = 0; i < 8; ++i) acc[nt][i] = 0.0f;

#pragma unroll 1
    for (int hc = 0; hc < H_DIM; hc += 128) {
        // ===== GEMM1: h[16x16 per wave] = x @ w1[:, hc + wv*16 ..] =========
        v8f ch;
#pragma unroll
        for (int i = 0; i < 8; ++i) ch[i] = 0.0f;

        const bf16t* bcol = w1e + (size_t)(hc / 16 + wv) * 32 * 512;
#pragma unroll 2
        for (int ks32 = 0; ks32 < 32; ++ks32) {
            FragU a, b;
            a.q[0] = *(const v4u*)&xlds[nl * XP + ks32 * 32 + klo];
            a.q[1] = *(const v4u*)&xlds[nl * XP + ks32 * 32 + 16 + klo];
            const v4u* bp = (const v4u*)(bcol + (size_t)ks32 * 512 + lane * 16);
            b.q[0] = bp[0];
            b.q[1] = bp[1];
            __builtin_prefetch(bcol + (size_t)(ks32 + 4) * 512 + lane * 16, 0, 1);
            ch = __builtin_amdgcn_wmma_f32_16x16x32_bf16(
                     false, a.v, false, b.v, (short)0, ch, false, false);
        }

        // bias + relu -> hlds
        float bias1 = b1[(size_t)e * H_DIM + hc + wv * 16 + nl];
        __syncthreads();   // previous chunk's GEMM2 reads of hlds done
        {
            int col = wv * 16 + nl;
            int mb  = half * 8;
#pragma unroll
            for (int i = 0; i < 8; ++i) {
                float v = ch[i] + bias1;
                v = (v > 0.0f) ? v : 0.0f;
                hlds[(mb + i) * HP + col] = (bf16t)v;
            }
        }
        __syncthreads();

        // ===== GEMM2: acc[nt] += h @ w2[hc.., wave D-slice] ================
#pragma unroll 1
        for (int ks2 = 0; ks2 < 4; ++ks2) {
            int kstep2 = hc / 32 + ks2;
            FragU a2;
            a2.q[0] = *(const v4u*)&hlds[nl * HP + ks2 * 32 + klo];
            a2.q[1] = *(const v4u*)&hlds[nl * HP + ks2 * 32 + 16 + klo];
#pragma unroll
            for (int nt = 0; nt < 8; ++nt) {
                const v4u* bp2 = (const v4u*)(
                    w2e + ((size_t)(wv * 8 + nt) * 128 + kstep2) * 512 + lane * 16);
                FragU bb;
                bb.q[0] = bp2[0];
                bb.q[1] = bp2[1];
                __builtin_prefetch((const bf16t*)bp2 + 512, 0, 1);  // next kstep
                acc[nt] = __builtin_amdgcn_wmma_f32_16x16x32_bf16(
                              false, a2.v, false, bb.v, (short)0, acc[nt],
                              false, false);
            }
        }
    }

    // ---- epilogue: out[t] += gate * (y + b2) ----
    float gm[8]; int tm[8];
    {
        int mb = half * 8;
#pragma unroll
        for (int i = 0; i < 8; ++i) {
            int r = tile * 16 + mb + i;
            if (r < ntok) {
                tm[i] = tlist[e * T_TOK + r];
                gm[i] = glist[e * T_TOK + r];
            } else { tm[i] = 0; gm[i] = 0.0f; }
        }
    }
#pragma unroll
    for (int nt = 0; nt < 8; ++nt) {
        int d = wv * 128 + nt * 16 + nl;
        float bv = b2[(size_t)e * D_EMB + d];
#pragma unroll
        for (int i = 0; i < 8; ++i) {
            float val = (acc[nt][i] + bv) * gm[i];
            atomicAdd(&out[(size_t)tm[i] * D_EMB + d], val);
        }
    }
}

// ---------------------------------------- ffn fallback (staged weights) ----
__global__ __launch_bounds__(256) void k_ffn(
    const float* __restrict__ x,  const float* __restrict__ w1,
    const float* __restrict__ b1, const float* __restrict__ w2,
    const float* __restrict__ b2, const int* __restrict__ cnt,
    const int* __restrict__ tlist, const float* __restrict__ glist,
    float* __restrict__ out) {

    __shared__ __attribute__((aligned(16))) bf16t xlds[16 * XP];
    __shared__ __attribute__((aligned(16))) bf16t bt1 [128 * BP];
    __shared__ __attribute__((aligned(16))) bf16t hlds[16 * HP];
    __shared__ __attribute__((aligned(16))) bf16t bt2 [8 * 16 * BP];

    int e    = blockIdx.x >> 8;
    int tile = blockIdx.x & 255;
    int ntok = cnt[e];
    if (tile * 16 >= ntok) return;

    int tid  = threadIdx.x;
    int lane = tid & 31, wv = tid >> 5;
    int half = lane >> 4;
    int klo  = half * 8;
    int nl   = lane & 15;

    const float* w1e = w1 + (size_t)e * D_EMB * H_DIM;
    const float* w2e = w2 + (size_t)e * H_DIM * D_EMB;

    {
        int m = tid >> 4, g = tid & 15;
        int r = tile * 16 + m;
        int rr = (r < ntok) ? r : (ntok - 1);
        int tok = tlist[e * T_TOK + rr];
        const float* xr = x + (size_t)tok * D_EMB;
#pragma unroll
        for (int q = 0; q < 16; ++q) {
            int col = q * 64 + g * 4;
            float4 v = *(const float4*)(xr + col);
            bf16t* dst = &xlds[m * XP + col];
            dst[0] = (bf16t)v.x; dst[1] = (bf16t)v.y;
            dst[2] = (bf16t)v.z; dst[3] = (bf16t)v.w;
        }
    }
    __syncthreads();

    v8f acc[8];
#pragma unroll
    for (int nt = 0; nt < 8; ++nt)
#pragma unroll
        for (int i = 0; i < 8; ++i) acc[nt][i] = 0.0f;

#pragma unroll 1
    for (int hc = 0; hc < H_DIM; hc += 128) {
        v8f ch;
#pragma unroll
        for (int i = 0; i < 8; ++i) ch[i] = 0.0f;

#pragma unroll 1
        for (int ks = 0; ks < D_EMB; ks += 32) {
            __syncthreads();
            {
                int kk = tid >> 3, g = tid & 7;
                const float* src = w1e + (size_t)(ks + kk) * H_DIM + hc + g * 16;
#pragma unroll
                for (int q = 0; q < 4; ++q) {
                    float4 v = *(const float4*)(src + q * 4);
                    int n0 = g * 16 + q * 4;
                    bt1[(n0 + 0) * BP + kk] = (bf16t)v.x;
                    bt1[(n0 + 1) * BP + kk] = (bf16t)v.y;
                    bt1[(n0 + 2) * BP + kk] = (bf16t)v.z;
                    bt1[(n0 + 3) * BP + kk] = (bf16t)v.w;
                }
            }
            __syncthreads();
            FragU a, b;
            a.q[0] = *(const v4u*)&xlds[nl * XP + ks + klo];
            a.q[1] = *(const v4u*)&xlds[nl * XP + ks + 16 + klo];
            int nb = wv * 16 + nl;
            b.q[0] = *(const v4u*)&bt1[nb * BP + klo];
            b.q[1] = *(const v4u*)&bt1[nb * BP + 16 + klo];
            ch = __builtin_amdgcn_wmma_f32_16x16x32_bf16(
                     false, a.v, false, b.v, (short)0, ch, false, false);
        }

        float bias1 = b1[(size_t)e * H_DIM + hc + wv * 16 + nl];
        __syncthreads();
        {
            int col = wv * 16 + nl;
            int mb  = half * 8;
#pragma unroll
            for (int i = 0; i < 8; ++i) {
                float v = ch[i] + bias1;
                v = (v > 0.0f) ? v : 0.0f;
                hlds[(mb + i) * HP + col] = (bf16t)v;
            }
        }
        __syncthreads();

#pragma unroll 1
        for (int ks2 = 0; ks2 < 128; ks2 += 32) {
            FragU a2;
            a2.q[0] = *(const v4u*)&hlds[nl * HP + ks2 + klo];
            a2.q[1] = *(const v4u*)&hlds[nl * HP + ks2 + 16 + klo];
            bf16t* st = &bt2[wv * (16 * BP)];
#pragma unroll 1
            for (int nt = 0; nt < 8; ++nt) {
                int dbase = wv * 128 + nt * 16;
                const float* src =
                    w2e + (size_t)(hc + ks2 + lane) * D_EMB + dbase;
                asm volatile("" ::: "memory");
#pragma unroll
                for (int q = 0; q < 4; ++q) {
                    float4 v = *(const float4*)(src + q * 4);
                    int n0 = q * 4;
                    st[(n0 + 0) * BP + lane] = (bf16t)v.x;
                    st[(n0 + 1) * BP + lane] = (bf16t)v.y;
                    st[(n0 + 2) * BP + lane] = (bf16t)v.z;
                    st[(n0 + 3) * BP + lane] = (bf16t)v.w;
                }
                asm volatile("" ::: "memory");
                FragU bf;
                bf.q[0] = *(const v4u*)&st[nl * BP + klo];
                bf.q[1] = *(const v4u*)&st[nl * BP + 16 + klo];
                acc[nt] = __builtin_amdgcn_wmma_f32_16x16x32_bf16(
                              false, a2.v, false, bf.v, (short)0, acc[nt],
                              false, false);
            }
        }
    }

    float gm[8]; int tm[8];
    {
        int mb = half * 8;
#pragma unroll
        for (int i = 0; i < 8; ++i) {
            int r = tile * 16 + mb + i;
            if (r < ntok) {
                tm[i] = tlist[e * T_TOK + r];
                gm[i] = glist[e * T_TOK + r];
            } else { tm[i] = 0; gm[i] = 0.0f; }
        }
    }
#pragma unroll
    for (int nt = 0; nt < 8; ++nt) {
        int d = wv * 128 + nt * 16 + nl;
        float bv = b2[(size_t)e * D_EMB + d];
#pragma unroll
        for (int i = 0; i < 8; ++i) {
            float val = (acc[nt][i] + bv) * gm[i];
            atomicAdd(&out[(size_t)tm[i] * D_EMB + d], val);
        }
    }
}

// -------------------------------------------------------------- launch ----
extern "C" void kernel_launch(void* const* d_in, const int* in_sizes, int n_in,
                              void* d_out, int out_size, void* d_ws,
                              size_t ws_size, hipStream_t stream) {
    const float* x  = (const float*)d_in[0];
    const float* rw = (const float*)d_in[1];
    const float* rb = (const float*)d_in[2];
    const float* w1 = (const float*)d_in[3];
    const float* b1 = (const float*)d_in[4];
    const float* w2 = (const float*)d_in[5];
    const float* b2 = (const float*)d_in[6];
    float* out = (float*)d_out;

    char* ws = (char*)d_ws;
    int*   cnt   = (int*)ws;
    int*   tlist = (int*)(ws + 1024);
    float* glist = (float*)(ws + 1024 + sizeof(int) * N_EXP * T_TOK);

    const size_t OFF_XB = (size_t)1 << 20;                       // 1 MB
    const size_t OFF_W1 = (size_t)16 << 20;                      // 16 MB
    const size_t SZ_W   = (size_t)N_EXP * D_EMB * H_DIM * 2;     // 64 MB
    const size_t OFF_W2 = OFF_W1 + SZ_W;
    const size_t NEED   = OFF_W2 + SZ_W;                         // 144 MB

    k_zero  <<<1024, 256, 0, stream>>>(out, cnt, T_TOK * D_EMB);
    k_router<<<T_TOK / 8, 256, 0, stream>>>(x, rw, rb, cnt, tlist, glist);

    if (ws_size >= NEED) {
        bf16t* xb  = (bf16t*)(ws + OFF_XB);
        bf16t* w1p = (bf16t*)(ws + OFF_W1);
        bf16t* w2p = (bf16t*)(ws + OFF_W2);
        k_pack_x<<<(T_TOK * D_EMB / 4 + 255) / 256, 256, 0, stream>>>(
            x, xb, T_TOK * D_EMB);
        k_pack_b<<<8192, 256, 0, stream>>>(w1, w1p, D_EMB, H_DIM, N_EXP);
        k_pack_b<<<8192, 256, 0, stream>>>(w2, w2p, H_DIM, D_EMB, N_EXP);
        k_ffn_packed<<<N_EXP * 256, 256, 0, stream>>>(
            xb, w1p, w2p, b1, b2, cnt, tlist, glist, out);
    } else {
        k_ffn<<<N_EXP * 256, 256, 0, stream>>>(x, w1, b1, w2, b2,
                                               cnt, tlist, glist, out);
    }
}
